// GILRLayer_67980742362008
// MI455X (gfx1250) — matrix operand order repaired and checked by reading, not compiled, not run on hardware
//
#include <hip/hip_runtime.h>
#include <math.h>

// GILR layer for MI455X (gfx1250, wave32).
// B=4, T=4096, Din=D=1024. Five 16384x1024x1024 GEMMs on the bf16 WMMA pipe
// (v_wmma_f32_16x16x32_bf16, f32 accumulation), chunked-affine parallel scan,
// fused activation epilogues, separate LayerNorm pass.
// Each wave computes a 32x64 C tile (8 WMMAs / 12 b128 loads per k-step,
// B fragments reused across two M rows). Requires ~236 MB of workspace.

#define B_    4
#define T_    4096
#define DIN   1024
#define D_    1024
#define M_    (B_ * T_)      // 16384 rows
#define CHUNK 128
#define NCH   (T_ / CHUNK)   // 32 chunks

typedef __attribute__((ext_vector_type(16))) __bf16 v16bf;
typedef __attribute__((ext_vector_type(8)))  __bf16 v8bf;
typedef __attribute__((ext_vector_type(8)))  float  v8f;

// ---- f32 -> bf16, round-to-nearest-even ----
__device__ __forceinline__ __bf16 f2bf(float f) {
  unsigned u = __builtin_bit_cast(unsigned, f);
  u += 0x7FFFu + ((u >> 16) & 1u);
  unsigned short s = (unsigned short)(u >> 16);
  return __builtin_bit_cast(__bf16, s);
}

// =====================================================================
// Elementwise f32 -> bf16 cast (for x)
// =====================================================================
__global__ void k_cast_bf16(const float* __restrict__ src,
                            __bf16* __restrict__ dst, int n) {
  int i = blockIdx.x * blockDim.x + threadIdx.x;
  if (i < n) dst[i] = f2bf(src[i]);
}

// =====================================================================
// Weight transpose + cast: W[K][N] f32 -> WT[N][K] bf16 (LDS-tiled)
// =====================================================================
__global__ void k_transpose_bf(const float* __restrict__ W,
                               __bf16* __restrict__ WT, int K, int N) {
  __shared__ __bf16 tile[32][33];
  const int k0 = blockIdx.y * 32;
  const int n0 = blockIdx.x * 32;
  for (int r = threadIdx.y; r < 32; r += 8)
    tile[r][threadIdx.x] = f2bf(W[(size_t)(k0 + r) * N + n0 + threadIdx.x]);
  __syncthreads();
  for (int r = threadIdx.y; r < 32; r += 8)
    WT[(size_t)(n0 + r) * K + k0 + threadIdx.x] = tile[threadIdx.x][r];
}

// =====================================================================
// WMMA core: one wave computes a 32x64 strip of C = A(32xK) * B(Kx64).
// A   : row-major [M][K] bf16
// BT  : row-major [N][K] bf16  (B transposed, so B-fragment loads are
//       contiguous along K, in the same k order as the A fragments)
// Layouts used here (wave32; A/B use contiguous k, not the ISA table order):
//   A 16x32: lane = half*16 + m ; elems 0..15 -> K = k + half*16 + j
//   B 32x16: lane = half*16 + n ; elems 0..15 -> K = k + half*16 + j
//   C 16x16 f32: lane = half*16 + n ; vgpr r -> M = half*8 + r
// Each B fragment is reused for both M sub-tiles: 8 WMMAs per 12 loads.
// =====================================================================
__device__ __forceinline__ void wmma_core32(const __bf16* __restrict__ A,
                                            const __bf16* __restrict__ BT,
                                            int K, int m0, int n0, int lane,
                                            v8f acc[2][4]) {
  const int half = lane >> 4;
  const int l15  = lane & 15;
  const __bf16* arow0 = A + (size_t)(m0 + l15) * K + half * 16;
  const __bf16* arow1 = arow0 + (size_t)16 * K;
  const v8f z = {0.f, 0.f, 0.f, 0.f, 0.f, 0.f, 0.f, 0.f};
#pragma unroll
  for (int mt = 0; mt < 2; ++mt)
#pragma unroll
    for (int t = 0; t < 4; ++t) acc[mt][t] = z;

  for (int k = 0; k < K; k += 32) {
    v8bf a0lo = *(const v8bf*)(arow0 + k);
    v8bf a0hi = *(const v8bf*)(arow0 + k + 8);
    v8bf a1lo = *(const v8bf*)(arow1 + k);
    v8bf a1hi = *(const v8bf*)(arow1 + k + 8);
    v16bf a0 = __builtin_shufflevector(a0lo, a0hi, 0, 1, 2, 3, 4, 5, 6, 7,
                                       8, 9, 10, 11, 12, 13, 14, 15);
    v16bf a1 = __builtin_shufflevector(a1lo, a1hi, 0, 1, 2, 3, 4, 5, 6, 7,
                                       8, 9, 10, 11, 12, 13, 14, 15);
#pragma unroll
    for (int t = 0; t < 4; ++t) {
      const __bf16* bp = BT + (size_t)(n0 + t * 16 + l15) * K + half * 16 + k;
      v16bf b = *(const v16bf*)bp;
      acc[0][t] = __builtin_amdgcn_wmma_f32_16x16x32_bf16(
          false, a0, false, b, (short)0, acc[0][t], false, false);
      acc[1][t] = __builtin_amdgcn_wmma_f32_16x16x32_bf16(
          false, a1, false, b, (short)0, acc[1][t], false, false);
    }
  }
}

// =====================================================================
// GEMM 1: gates.  e = blockIdx.z selects Win[0] (tanh -> V) or
// Win[1] (sigmoid * (1 - rnn_start) -> F).
// =====================================================================
__global__ void k_gemm_gate(const __bf16* __restrict__ XB,
                            const __bf16* __restrict__ WT0,
                            const __bf16* __restrict__ WT1,
                            const float* __restrict__ bin,
                            const float* __restrict__ rnn_start,
                            float* __restrict__ V, float* __restrict__ F) {
  const int lane = threadIdx.x & 31;
  const int wave = threadIdx.x >> 5;
  const int m0 = blockIdx.y * 256 + wave * 32;
  const int n0 = blockIdx.x * 64;
  const int e  = blockIdx.z;
  const __bf16* BT = e ? WT1 : WT0;
  v8f acc[2][4];
  wmma_core32(XB, BT, DIN, m0, n0, lane, acc);

  const int half = lane >> 4, l15 = lane & 15;
  const float* bias = bin + e * D_;
#pragma unroll
  for (int mt = 0; mt < 2; ++mt) {
#pragma unroll
    for (int t = 0; t < 4; ++t) {
      const int col = n0 + t * 16 + l15;
      const float bb = bias[col];
#pragma unroll
      for (int r = 0; r < 8; ++r) {
        const int row = m0 + mt * 16 + half * 8 + r;
        const float u = acc[mt][t][r] + bb;
        if (e == 0) {
          V[(size_t)row * D_ + col] = tanhf(u);
        } else {
          const float s = 1.f / (1.f + expf(-u));
          F[(size_t)row * D_ + col] = s * (1.f - rnn_start[row]);
        }
      }
    }
  }
}

// =====================================================================
// Chunked affine scan: h_t = f_t*h + (1-f_t)*v_t
//   phase 1: per-(b,chunk,d) compose 128 affine maps -> (A, C)
//   phase 2: per-(b,d) sequential over 32 chunks; emits chunk-start h
//            and final hidden state
//   phase 3: per-(b,chunk,d) replay chunk, emit hs as bf16 for next GEMM
// d is the fastest thread index everywhere -> coalesced.
// =====================================================================
__global__ void k_scan_phase1(const float* __restrict__ F,
                              const float* __restrict__ V,
                              float* __restrict__ CA, float* __restrict__ CC) {
  const int tid = blockIdx.x * blockDim.x + threadIdx.x;  // B*NCH*D
  const int d = tid & (D_ - 1);
  const int rest = tid >> 10;
  const int ch = rest & (NCH - 1);
  const int b = rest >> 5;
  const size_t base = ((size_t)(b * T_ + ch * CHUNK) << 10) + d;
  float A = 1.f, C = 0.f;
  for (int t = 0; t < CHUNK; ++t) {
    const float f = F[base + ((size_t)t << 10)];
    const float v = V[base + ((size_t)t << 10)];
    A = f * A;
    C = f * C + (1.f - f) * v;
  }
  CA[tid] = A;
  CC[tid] = C;
}

__global__ void k_scan_phase2(const float* __restrict__ hidden,
                              const float* __restrict__ CA,
                              const float* __restrict__ CC,
                              float* __restrict__ HST,
                              float* __restrict__ out_hidden) {
  const int tid = blockIdx.x * blockDim.x + threadIdx.x;  // B*D
  const int d = tid & (D_ - 1);
  const int b = tid >> 10;
  float h = hidden[tid];  // hidden is (B,1,D): index b*D + d == tid
  for (int ch = 0; ch < NCH; ++ch) {
    const size_t i = ((size_t)(b * NCH + ch) << 10) + d;
    HST[i] = h;
    h = CA[i] * h + CC[i];
  }
  out_hidden[tid] = h;  // hidden_new, appended after `out` in d_out
}

__global__ void k_scan_phase3(const float* __restrict__ F,
                              const float* __restrict__ V,
                              const float* __restrict__ HST,
                              __bf16* __restrict__ HSB) {
  const int tid = blockIdx.x * blockDim.x + threadIdx.x;  // B*NCH*D
  const int d = tid & (D_ - 1);
  const int rest = tid >> 10;
  const int ch = rest & (NCH - 1);
  const int b = rest >> 5;
  const size_t base = ((size_t)(b * T_ + ch * CHUNK) << 10) + d;
  float h = HST[tid];
  for (int t = 0; t < CHUNK; ++t) {
    const size_t i = base + ((size_t)t << 10);
    const float f = F[i];
    h = f * h + (1.f - f) * V[i];
    HSB[i] = f2bf(h);
  }
}

// =====================================================================
// GEMM 2: out1 = hs @ Wout + bout  (f32 for residual, bf16 for next GEMM)
// =====================================================================
__global__ void k_gemm_out(const __bf16* __restrict__ HSB,
                           const __bf16* __restrict__ WT,
                           const float* __restrict__ bout,
                           float* __restrict__ OUT1F,
                           __bf16* __restrict__ OUT1B) {
  const int lane = threadIdx.x & 31;
  const int wave = threadIdx.x >> 5;
  const int m0 = blockIdx.y * 256 + wave * 32;
  const int n0 = blockIdx.x * 64;
  v8f acc[2][4];
  wmma_core32(HSB, WT, D_, m0, n0, lane, acc);
  const int half = lane >> 4, l15 = lane & 15;
#pragma unroll
  for (int mt = 0; mt < 2; ++mt) {
#pragma unroll
    for (int t = 0; t < 4; ++t) {
      const int col = n0 + t * 16 + l15;
      const float bb = bout[col];
#pragma unroll
      for (int r = 0; r < 8; ++r) {
        const size_t idx = (size_t)(m0 + mt * 16 + half * 8 + r) * D_ + col;
        const float u = acc[mt][t][r] + bb;
        OUT1F[idx] = u;
        OUT1B[idx] = f2bf(u);
      }
    }
  }
}

// =====================================================================
// GEMM 3: x_ = gelu_exact(out1 @ W1 + b1)
// =====================================================================
__global__ void k_gemm_gelu(const __bf16* __restrict__ O1B,
                            const __bf16* __restrict__ WT,
                            const float* __restrict__ b1,
                            __bf16* __restrict__ XGB) {
  const int lane = threadIdx.x & 31;
  const int wave = threadIdx.x >> 5;
  const int m0 = blockIdx.y * 256 + wave * 32;
  const int n0 = blockIdx.x * 64;
  v8f acc[2][4];
  wmma_core32(O1B, WT, D_, m0, n0, lane, acc);
  const int half = lane >> 4, l15 = lane & 15;
#pragma unroll
  for (int mt = 0; mt < 2; ++mt) {
#pragma unroll
    for (int t = 0; t < 4; ++t) {
      const int col = n0 + t * 16 + l15;
      const float bb = b1[col];
#pragma unroll
      for (int r = 0; r < 8; ++r) {
        const size_t idx = (size_t)(m0 + mt * 16 + half * 8 + r) * D_ + col;
        const float u = acc[mt][t][r] + bb;
        const float g = 0.5f * u * (1.f + erff(u * 0.70710678118654752f));
        XGB[idx] = f2bf(g);
      }
    }
  }
}

// =====================================================================
// GEMM 4: y = x_ @ W2 + b2 + out1   (pre-LayerNorm, f32)
// =====================================================================
__global__ void k_gemm_res(const __bf16* __restrict__ XGB,
                           const __bf16* __restrict__ WT,
                           const float* __restrict__ b2,
                           const float* __restrict__ OUT1F,
                           float* __restrict__ Y) {
  const int lane = threadIdx.x & 31;
  const int wave = threadIdx.x >> 5;
  const int m0 = blockIdx.y * 256 + wave * 32;
  const int n0 = blockIdx.x * 64;
  v8f acc[2][4];
  wmma_core32(XGB, WT, D_, m0, n0, lane, acc);
  const int half = lane >> 4, l15 = lane & 15;
#pragma unroll
  for (int mt = 0; mt < 2; ++mt) {
#pragma unroll
    for (int t = 0; t < 4; ++t) {
      const int col = n0 + t * 16 + l15;
      const float bb = b2[col];
#pragma unroll
      for (int r = 0; r < 8; ++r) {
        const size_t idx = (size_t)(m0 + mt * 16 + half * 8 + r) * D_ + col;
        Y[idx] = acc[mt][t][r] + bb + OUT1F[idx];
      }
    }
  }
}

// =====================================================================
// LayerNorm over D=1024, one block (256 threads) per row
// =====================================================================
__global__ void k_layernorm(const float* __restrict__ Y,
                            const float* __restrict__ g,
                            const float* __restrict__ bb,
                            float* __restrict__ out) {
  __shared__ float s1[256], s2[256];
  const int row = blockIdx.x;
  const float* y = Y + (size_t)row * D_;
  float a = 0.f, q = 0.f;
  for (int i = threadIdx.x; i < D_; i += 256) {
    const float v = y[i];
    a += v;
    q += v * v;
  }
  s1[threadIdx.x] = a;
  s2[threadIdx.x] = q;
  __syncthreads();
  for (int s = 128; s > 0; s >>= 1) {
    if (threadIdx.x < s) {
      s1[threadIdx.x] += s1[threadIdx.x + s];
      s2[threadIdx.x] += s2[threadIdx.x + s];
    }
    __syncthreads();
  }
  const float mean = s1[0] * (1.f / D_);
  const float var  = s2[0] * (1.f / D_) - mean * mean;
  const float inv  = rsqrtf(var + 1e-5f);
  for (int i = threadIdx.x; i < D_; i += 256)
    out[(size_t)row * D_ + i] = (y[i] - mean) * inv * g[i] + bb[i];
}

// =====================================================================
extern "C" void kernel_launch(void* const* d_in, const int* in_sizes, int n_in,
                              void* d_out, int out_size, void* d_ws,
                              size_t ws_size, hipStream_t stream) {
  (void)in_sizes; (void)n_in; (void)out_size; (void)ws_size;

  const float* x         = (const float*)d_in[0];
  const float* hidden    = (const float*)d_in[1];
  const float* rnn_start = (const float*)d_in[2];
  const float* Win       = (const float*)d_in[3];
  const float* bin       = (const float*)d_in[4];
  const float* Wout      = (const float*)d_in[5];
  const float* bout      = (const float*)d_in[6];
  const float* W1        = (const float*)d_in[7];
  const float* b1        = (const float*)d_in[8];
  const float* W2        = (const float*)d_in[9];
  const float* b2        = (const float*)d_in[10];
  const float* lng       = (const float*)d_in[11];
  const float* lnb       = (const float*)d_in[12];
  float* out = (float*)d_out;
  char* ws = (char*)d_ws;

  const size_t MB = (size_t)1 << 20;
  // Workspace layout (~236 MB, with reuse of dead buffers):
  __bf16* XB    = (__bf16*)(ws + 0);          // 32 MB  x as bf16
  __bf16* WT0   = (__bf16*)(ws + 32 * MB);    //  2 MB  Win[0]^T
  __bf16* WT1   = (__bf16*)(ws + 34 * MB);    //  2 MB  Win[1]^T
  __bf16* WOT   = (__bf16*)(ws + 36 * MB);    //  2 MB  Wout^T
  __bf16* W1T   = (__bf16*)(ws + 38 * MB);    //  2 MB  W1^T
  __bf16* W2T   = (__bf16*)(ws + 40 * MB);    //  2 MB  W2^T
  float*  CA    = (float*)(ws + 42 * MB);     // 512 KB chunk A
  float*  CC    = (float*)(ws + 42 * MB + 512 * 1024); // 512 KB chunk C
  float*  HST   = (float*)(ws + 43 * MB);     // 512 KB chunk-start h
  float*  V     = (float*)(ws + 44 * MB);     // 64 MB  tanh gate
  float*  F     = (float*)(ws + 108 * MB);    // 64 MB  sigmoid gate
  __bf16* HSB   = (__bf16*)(ws + 172 * MB);   // 32 MB  hs bf16
  float*  OUT1F = (float*)(ws + 44 * MB);     // reuse V  (dead after scan)
  __bf16* OUT1B = (__bf16*)(ws + 108 * MB);   // reuse F  (dead after scan)
  __bf16* XGB   = (__bf16*)(ws + 140 * MB);   // 32 MB  gelu output bf16
  float*  Y     = (float*)(ws + 172 * MB);    // reuse HSB (+32MB) -> 64 MB

  const dim3 blk(256);

  // 1. cast x to bf16
  k_cast_bf16<<<dim3((M_ * DIN + 255) / 256), blk, 0, stream>>>(x, XB, M_ * DIN);

  // 2. transpose + cast all weights
  const dim3 tb(32, 8);
  const dim3 tg(D_ / 32, DIN / 32);
  k_transpose_bf<<<tg, tb, 0, stream>>>(Win,            WT0, DIN, D_);
  k_transpose_bf<<<tg, tb, 0, stream>>>(Win + DIN * D_, WT1, DIN, D_);
  k_transpose_bf<<<tg, tb, 0, stream>>>(Wout, WOT, D_, D_);
  k_transpose_bf<<<tg, tb, 0, stream>>>(W1,   W1T, D_, D_);
  k_transpose_bf<<<tg, tb, 0, stream>>>(W2,   W2T, D_, D_);

  // 3. gate GEMMs (z-dim picks tanh/sigmoid branch)
  k_gemm_gate<<<dim3(D_ / 64, M_ / 256, 2), blk, 0, stream>>>(
      XB, WT0, WT1, bin, rnn_start, V, F);

  // 4. chunked affine scan
  k_scan_phase1<<<dim3(B_ * NCH * D_ / 256), blk, 0, stream>>>(F, V, CA, CC);
  k_scan_phase2<<<dim3(B_ * D_ / 256), blk, 0, stream>>>(
      hidden, CA, CC, HST, out + (size_t)M_ * D_);
  k_scan_phase3<<<dim3(B_ * NCH * D_ / 256), blk, 0, stream>>>(F, V, HST, HSB);

  // 5. output projection + MLP + residual
  const dim3 gg(D_ / 64, M_ / 256);
  k_gemm_out <<<gg, blk, 0, stream>>>(HSB, WOT, bout, OUT1F, OUT1B);
  k_gemm_gelu<<<gg, blk, 0, stream>>>(OUT1B, W1T, b1, XGB);
  k_gemm_res <<<gg, blk, 0, stream>>>(XGB, W2T, b2, OUT1F, Y);

  // 6. LayerNorm -> final output
  k_layernorm<<<dim3(M_), blk, 0, stream>>>(Y, lng, lnb, out);
}
